// CgpHmmLayer_59949153517628
// MI455X (gfx1250) — matrix operand order repaired and checked by reading, not compiled, not run on hardware
//
#include <hip/hip_runtime.h>
#include <hip/hip_bf16.h>
#include <math.h>

// Problem constants (from reference)
#define S_DIM 610
#define E_DIM 126
#define BATCH 32
#define T_LEN 2048
#define NREG  5000
#define ALPHA_REG 4.0f

// Padded dims for WMMA tiling
#define SP 640            // K (state) dim padded to multiple of 32
#define NP 624            // N (state) dim padded to 39*16
#define NTILES 39

typedef __attribute__((ext_vector_type(16))) __bf16 v16bf;
typedef __attribute__((ext_vector_type(8)))  __bf16 v8bf;
typedef __attribute__((ext_vector_type(8)))  float  v8f;

__device__ __forceinline__ unsigned short f2bf(float f) {
    unsigned int u = __float_as_uint(f);
    unsigned int r = (u + 0x7FFFu + ((u >> 16) & 1u)) >> 16;   // round-to-nearest-even
    return (unsigned short)r;
}

// ---------------------------------------------------------------------------
// 1) Recover integer observations from one-hot inputs: obs[b*T+t] = argmax_e
// ---------------------------------------------------------------------------
__global__ void obs_kernel(const float* __restrict__ inputs, int* __restrict__ obs) {
    int idx = blockIdx.x * blockDim.x + threadIdx.x;     // flat b*T + t
    if (idx >= BATCH * T_LEN) return;
    const float* row = inputs + (size_t)idx * E_DIM;
    int best = 0; float bv = row[0];
    for (int e = 1; e < E_DIM; ++e) {
        float v = row[e];
        if (v > bv) { bv = v; best = e; }
    }
    obs[idx] = best;
}

// ---------------------------------------------------------------------------
// 2) Pack transition matrix A (row-major [from k][to n], fp32) into bf16
//    column-major Apk[n*SP + k], zero-padded.  This makes each lane's WMMA
//    B-operand fragment a contiguous 32B run along K.
// ---------------------------------------------------------------------------
__global__ void pack_kernel(const float* __restrict__ A, unsigned short* __restrict__ Apk) {
    int idx = blockIdx.x * blockDim.x + threadIdx.x;
    if (idx >= NP * SP) return;
    int n = idx / SP;
    int k = idx - n * SP;
    float v = (n < S_DIM && k < S_DIM) ? A[(size_t)k * S_DIM + n] : 0.0f;
    Apk[idx] = f2bf(v);
}

// ---------------------------------------------------------------------------
// 3) t=0: alpha0 = I * Bm[:, obs[b,0]], normalize, ll0 = log z0
//    One block per batch; writes bf16 alpha0 [b][SP] and ll[b].
// ---------------------------------------------------------------------------
__global__ void init_kernel(const float* __restrict__ I, const float* __restrict__ Bm,
                            const int* __restrict__ obs,
                            unsigned short* __restrict__ alpha0, float* __restrict__ ll) {
    __shared__ float red[256];
    __shared__ float zsh;
    int b = blockIdx.x, tid = threadIdx.x;
    int o = obs[b * T_LEN];
    float loc[3];
    float s = 0.f;
    #pragma unroll
    for (int i = 0; i < 3; ++i) {
        int st = tid + i * 256;
        float v = (st < S_DIM) ? I[st] * Bm[(size_t)st * E_DIM + o] : 0.f;
        loc[i] = v; s += v;
    }
    red[tid] = s; __syncthreads();
    for (int off = 128; off > 0; off >>= 1) {
        if (tid < off) red[tid] += red[tid + off];
        __syncthreads();
    }
    if (tid == 0) { zsh = red[0]; ll[b] = logf(red[0]); }
    __syncthreads();
    float rz = 1.f / zsh;
    #pragma unroll
    for (int i = 0; i < 3; ++i) {
        int st = tid + i * 256;
        if (st < SP) alpha0[(size_t)b * SP + st] = f2bf((st < S_DIM) ? loc[i] * rz : 0.f);
    }
}

// ---------------------------------------------------------------------------
// 4) Persistent forward scan.  Grid = 2 blocks x 256 threads.
//    Block g owns batches [16g, 16g+16).  Alpha lives in LDS (bf16),
//    per-step GEMM [16 x 640] x [640 x 624] via v_wmma_f32_16x16x32_bf16.
// ---------------------------------------------------------------------------
__launch_bounds__(256, 1)
__global__ void forward_kernel(const unsigned short* __restrict__ Apk,
                               const float* __restrict__ Bm,
                               const int* __restrict__ obs,
                               const unsigned short* __restrict__ alpha0,
                               float* __restrict__ ll) {
    __shared__ unsigned short sAlpha[16 * SP];   // 20 KB, bf16 alpha (normalized)
    __shared__ int   sObs[16];
    __shared__ float sSum[16];
    __shared__ float sRnorm[16];
    __shared__ float sLL[16];

    const int g     = blockIdx.x;
    const int tid   = threadIdx.x;
    const int wave  = tid >> 5;
    const int lane  = tid & 31;
    const int hi    = lane >> 4;          // lane half: 0 or 1
    const int lm    = lane & 15;          // position within half
    const int mBase = hi * 8;             // C/D rows owned by this lane half

    // ---- init: alpha0 -> LDS, ll, obs(t=1), zero sums ----
    for (int i = tid; i < 16 * SP; i += 256)
        sAlpha[i] = alpha0[(size_t)(g * 16) * SP + i];
    if (tid < 16) {
        sLL[tid]  = ll[g * 16 + tid];
        sSum[tid] = 0.f;
        sObs[tid] = obs[(g * 16 + tid) * T_LEN + 1];
    }
    __syncthreads();

    for (int t = 1; t < T_LEN; ++t) {
        // emission symbol per batch row handled by this lane
        int o[8];
        #pragma unroll
        for (int r = 0; r < 8; ++r) o[r] = sObs[mBase + r];

        // A-operand source: row m = lm of sAlpha; lane half selects contiguous 16-K half
        const unsigned short* rowA = &sAlpha[lm * SP + (hi ? 16 : 0)];

        v8f vals[5];
        #pragma unroll
        for (int i = 0; i < 5; ++i) {
            int nt = wave + 8 * i;
            if (nt < NTILES) {
                const int nn = nt * 16 + lm;                      // global column (to-state)
                const unsigned short* colB = Apk + (size_t)nn * SP + (hi ? 16 : 0);
                __builtin_prefetch((const void*)colB, 0, 1);      // global_prefetch_b8

                v8f acc = {0.f, 0.f, 0.f, 0.f, 0.f, 0.f, 0.f, 0.f};
                for (int k0 = 0; k0 < SP; k0 += 32) {
                    // A frag: 16 contiguous K starting at k0 + (hi?16:0)
                    v8bf a0 = *(const v8bf*)(rowA + k0);
                    v8bf a1 = *(const v8bf*)(rowA + k0 + 8);
                    v16bf afrag = __builtin_shufflevector(a0, a1,
                        0,1,2,3,4,5,6,7, 8,9,10,11,12,13,14,15);
                    // B frag: 16 contiguous K starting at k0 + (hi?16:0)
                    v8bf b0 = *(const v8bf*)(colB + k0);
                    v8bf b1 = *(const v8bf*)(colB + k0 + 8);
                    v16bf bfrag = __builtin_shufflevector(b0, b1,
                        0,1,2,3,4,5,6,7, 8,9,10,11,12,13,14,15);
                    acc = __builtin_amdgcn_wmma_f32_16x16x32_bf16(
                        false, afrag, false, bfrag, (short)0, acc, false, false);
                }

                // multiply by emission, reduce over states (N) per batch (M)
                const bool valid = (nn < S_DIM);
                #pragma unroll
                for (int r = 0; r < 8; ++r) {
                    float em = valid ? Bm[(size_t)nn * E_DIM + o[r]] : 0.f;
                    float v  = acc[r] * em;
                    acc[r]   = v;
                    float red = v;
                    red += __shfl_xor(red, 1, 32);
                    red += __shfl_xor(red, 2, 32);
                    red += __shfl_xor(red, 4, 32);
                    red += __shfl_xor(red, 8, 32);     // reduces each 16-lane half
                    if (lm == 0) atomicAdd(&sSum[mBase + r], red);  // ds_add_f32
                }
                vals[i] = acc;
            }
        }

        __syncthreads();     // all partial sums landed

        if (tid < 16) {
            float z = sSum[tid];
            sLL[tid]   += logf(z);
            sRnorm[tid] = 1.f / z;
            sSum[tid]   = 0.f;
            int tn = (t + 1 < T_LEN) ? (t + 1) : (T_LEN - 1);
            sObs[tid] = obs[(g * 16 + tid) * T_LEN + tn];
        }
        __syncthreads();     // rnorm / next-obs ready; alpha reads done

        // write normalized bf16 alpha back to LDS for next step
        #pragma unroll
        for (int i = 0; i < 5; ++i) {
            int nt = wave + 8 * i;
            if (nt < NTILES) {
                const int nn = nt * 16 + lm;
                #pragma unroll
                for (int r = 0; r < 8; ++r) {
                    float rn = sRnorm[mBase + r];
                    sAlpha[(mBase + r) * SP + nn] = f2bf(vals[i][r] * rn);
                }
            }
        }
        __syncthreads();     // alpha ready for next step
    }

    if (tid < 16) ll[g * 16 + tid] = sLL[tid];
}

// ---------------------------------------------------------------------------
// 5) Loss: -mean(ll) - ALPHA_REG * mean(log1p(-A[f,t])) over NREG pairs
// ---------------------------------------------------------------------------
__global__ void loss_kernel(const float* __restrict__ A, const int* __restrict__ rf,
                            const int* __restrict__ rt, const float* __restrict__ ll,
                            float* __restrict__ out) {
    __shared__ float red[256];
    int tid = threadIdx.x;
    float s = 0.f;
    for (int i = tid; i < NREG; i += 256) {
        float a = A[(size_t)rf[i] * S_DIM + rt[i]];
        s += log1pf(-a);
    }
    red[tid] = s; __syncthreads();
    for (int off = 128; off > 0; off >>= 1) {
        if (tid < off) red[tid] += red[tid + off];
        __syncthreads();
    }
    if (tid == 0) {
        float lls = 0.f;
        for (int b = 0; b < BATCH; ++b) lls += ll[b];
        out[0] = -(lls / (float)BATCH) - ALPHA_REG * (red[0] / (float)NREG);
    }
}

// ---------------------------------------------------------------------------
// Launch
// ---------------------------------------------------------------------------
extern "C" void kernel_launch(void* const* d_in, const int* in_sizes, int n_in,
                              void* d_out, int out_size, void* d_ws, size_t ws_size,
                              hipStream_t stream) {
    const float* inputs = (const float*)d_in[0];   // [B,T,E] one-hot
    const float* A      = (const float*)d_in[1];   // [S,S]
    const float* Bm     = (const float*)d_in[2];   // [S,E]
    const float* Iv     = (const float*)d_in[3];   // [S]
    const int*   rf     = (const int*)d_in[4];     // [NREG]
    const int*   rt     = (const int*)d_in[5];     // [NREG]

    char* ws = (char*)d_ws;
    size_t off = 0;
    int*            obs    = (int*)(ws + off);            off += (size_t)BATCH * T_LEN * 4;   // 256 KB
    unsigned short* Apk    = (unsigned short*)(ws + off); off += (size_t)NP * SP * 2;         // 780 KB
    unsigned short* alpha0 = (unsigned short*)(ws + off); off += (size_t)BATCH * SP * 2;      // 40 KB
    float*          ll     = (float*)(ws + off);          off += (size_t)BATCH * 4;

    float* out = (float*)d_out;

    obs_kernel   <<<(BATCH * T_LEN + 255) / 256, 256, 0, stream>>>(inputs, obs);
    pack_kernel  <<<(NP * SP + 255) / 256,       256, 0, stream>>>(A, Apk);
    init_kernel  <<<BATCH,                       256, 0, stream>>>(Iv, Bm, obs, alpha0, ll);
    forward_kernel<<<2,                          256, 0, stream>>>(Apk, Bm, obs, alpha0, ll);
    loss_kernel  <<<1,                           256, 0, stream>>>(A, rf, rt, ll, out);
}
